// GaussianMoELayer_74629351735722
// MI455X (gfx1250) — matrix operand order repaired and checked by reading, not compiled, not run on hardware
//
#include <hip/hip_runtime.h>
#include <hip/hip_bf16.h>
#include <math.h>

typedef __attribute__((ext_vector_type(16))) _Float16 v16h;
typedef __attribute__((ext_vector_type(8)))  float    v8f;

#define E_    16
#define IN_   256
#define H_    1024
#define OUT_  256
#define NTOK  8192

#define XS 264    // x tile LDS row stride in halves (256 + 8 pad -> conflict-free b128)
#define HS 1032   // h tile LDS row stride in halves (1024 + 8 pad)
// per-wave LDS: 32 rows of x + 32 rows of h
#define WAVE_SMEM_HALVES (32*XS + 32*HS)   // 8448 + 33024 = 41472 halves (82,944 B)

// ---------------------------------------------------------------- zero output
__global__ void zero_out_kernel(float* __restrict__ p, int n) {
  int i = blockIdx.x * blockDim.x + threadIdx.x;
  if (i < n) p[i] = 0.0f;
}

// ------------------------------------------------- pack weights into B-frags
// W: [E][K][Nn] f32 row-major.  Fragment (e, nt, kt) covers rows kt*32..+31,
// cols nt*16..+15.  Contiguous k order (same as the A loads): lane<16 holds column
// nt*16+lane, K = kt*32+0..15 (consecutive halves); lane>=16 same column,
// K = kt*32+16..31.  Each lane's 16 halves stored contiguously (32 B).
__global__ __launch_bounds__(256)
void pack_w_kernel(const float* __restrict__ W, _Float16* __restrict__ dst,
                   int K, int Nn) {
  int gid  = blockIdx.x * blockDim.x + threadIdx.x;
  int lane = gid & 31;
  int frag = gid >> 5;
  int nkt  = K >> 5;
  int nnt  = Nn >> 4;
  int kt   = frag % nkt;
  int rest = frag / nkt;
  int nt   = rest % nnt;
  int e    = rest / nnt;
  if (e >= E_) return;
  int col   = nt * 16 + (lane & 15);
  int kbase = kt * 32 + (lane >> 4) * 16;
  union { _Float16 h[16]; uint4 u[2]; } v;
  const float* src = W + (size_t)(e * K + kbase) * Nn + col;
#pragma unroll
  for (int j = 0; j < 16; ++j) v.h[j] = (_Float16)src[(size_t)j * Nn];
  uint4* d = (uint4*)(dst + (size_t)frag * 512 + lane * 16);
  d[0] = v.u[0];
  d[1] = v.u[1];
}

// ------------------------------------------------------------------- routing
__global__ __launch_bounds__(256)
void routing_kernel(const float* __restrict__ x, const float* __restrict__ mus,
                    const float* __restrict__ logsig, float* __restrict__ wts,
                    float* __restrict__ logp, int* __restrict__ topi) {
  __shared__ float smu[E_ * IN_];
  __shared__ float sinv[E_ * IN_];
  __shared__ float ssum[E_];
  int tid = threadIdx.x;
  for (int i = tid; i < E_ * IN_; i += 256) {
    smu[i]  = mus[i];
    sinv[i] = expf(-logsig[i]);           // 1 / sigma
  }
  __syncthreads();
  if (tid < E_) {
    float s = 0.f;
    for (int i = 0; i < IN_; ++i) s += logsig[tid * IN_ + i];
    ssum[tid] = s;
  }
  __syncthreads();

  int n = blockIdx.x * 256 + tid;
  const float4* xp = (const float4*)(x + (size_t)n * IN_);
  float lp[E_];
  for (int e = 0; e < E_; ++e) {
    const float4* mp = (const float4*)(smu + e * IN_);
    const float4* ip = (const float4*)(sinv + e * IN_);
    float d = 0.f;
    for (int i = 0; i < IN_ / 4; ++i) {
      float4 xv = xp[i], mv = mp[i], iv = ip[i];
      float a0 = (xv.x - mv.x) * iv.x, a1 = (xv.y - mv.y) * iv.y;
      float a2 = (xv.z - mv.z) * iv.z, a3 = (xv.w - mv.w) * iv.w;
      d += a0 * a0 + a1 * a1 + a2 * a2 + a3 * a3;
    }
    lp[e] = -0.5f * d - ssum[e];
  }
  float m = lp[0];
  for (int e = 1; e < E_; ++e) m = fmaxf(m, lp[e]);
  float s = 0.f, w[E_];
  for (int e = 0; e < E_; ++e) { w[e] = expf(lp[e] - m); s += w[e]; }
  float inv = 1.f / s;
  for (int e = 0; e < E_; ++e) {
    wts [(size_t)n * E_ + e] = w[e] * inv;
    logp[(size_t)n * E_ + e] = lp[e];
  }
  int i1 = 0; float v1 = lp[0];
  for (int e = 1; e < E_; ++e) if (lp[e] > v1) { v1 = lp[e]; i1 = e; }
  int i2 = 0; float v2 = -3.4e38f;
  for (int e = 0; e < E_; ++e) if (e != i1 && lp[e] > v2) { v2 = lp[e]; i2 = e; }
  topi[n * 2]     = i1;
  topi[n * 2 + 1] = i2;
}

// --------------------------------------------------------- fused MoE GEMMs
// grid = (NTOK/64, E).  64 threads = 2 independent wave32s; each wave owns
// TWO 16-row blocks (32 tokens) for one expert, so every B fragment loaded
// feeds two WMMAs (2x arithmetic intensity on the weight stream).
// h = gelu(x@W1+b1) staged f16 in LDS, then out += w[token,e]*(h@W2+b2)
// accumulated across experts with f32 global atomics.
__global__ __launch_bounds__(64)
void moe_gemm_kernel(const float* __restrict__ x, const _Float16* __restrict__ w1p,
                     const _Float16* __restrict__ w2p, const float* __restrict__ b1,
                     const float* __restrict__ b2, const float* __restrict__ wts,
                     float* __restrict__ out) {
  extern __shared__ _Float16 smem[];
  const int wave  = threadIdx.x >> 5;
  const int lane  = threadIdx.x & 31;
  const int e     = blockIdx.y;
  const int tbase = blockIdx.x * 64 + wave * 32;
  _Float16* xs = smem + wave * WAVE_SMEM_HALVES;
  _Float16* hs = xs + 32 * XS;

  // stage this wave's x tile (32 x 256) as f16 in LDS
  for (int it = 0; it < 64; ++it) {
    int idx = it * 32 + lane;
    int row = idx >> 6;
    int c4  = (idx & 63) * 4;
    float4 v = *(const float4*)(x + (size_t)(tbase + row) * IN_ + c4);
    union { _Float16 h[4]; uint2 u; } p;
    p.h[0] = (_Float16)v.x; p.h[1] = (_Float16)v.y;
    p.h[2] = (_Float16)v.z; p.h[3] = (_Float16)v.w;
    *(uint2*)(xs + row * XS + c4) = p.u;
  }

  const int hi   = lane >> 4;   // A-frag K-half select; C row-half select
  const int lcol = lane & 15;   // A row / B,C column

  // ---- Phase 1: h = gelu(x @ W1 + b1) -> f16 in LDS
  for (int hc = 0; hc < 16; ++hc) {           // 64 H columns per chunk
    if (hc + 1 < 16)
      __builtin_prefetch(w1p + ((((size_t)e * 64 + (hc + 1) * 4) * 8) << 9) + lane * 16, 0, 3);
    v8f acc[2][4] = {};
    for (int kt = 0; kt < 8; ++kt) {          // K = 256
      union { uint4 u[2]; v16h v; } A[2];
#pragma unroll
      for (int rb = 0; rb < 2; ++rb) {
        const _Float16* ap = xs + (rb * 16 + lcol) * XS + kt * 32 + hi * 16;
        A[rb].u[0] = *(const uint4*)ap;
        A[rb].u[1] = *(const uint4*)(ap + 8);
      }
#pragma unroll
      for (int nt = 0; nt < 4; ++nt) {
        union { uint4 u[2]; v16h v; } Bf;
        const _Float16* bp = w1p + ((((size_t)e * 64 + (hc * 4 + nt)) * 8 + kt) << 9) + lane * 16;
        Bf.u[0] = *(const uint4*)bp;
        Bf.u[1] = *(const uint4*)(bp + 8);
        acc[0][nt] = __builtin_amdgcn_wmma_f32_16x16x32_f16(
            false, A[0].v, false, Bf.v, (short)0, acc[0][nt], false, false);
        acc[1][nt] = __builtin_amdgcn_wmma_f32_16x16x32_f16(
            false, A[1].v, false, Bf.v, (short)0, acc[1][nt], false, false);
      }
    }
#pragma unroll
    for (int nt = 0; nt < 4; ++nt) {
      int col = hc * 64 + nt * 16 + lcol;
      float b1v = b1[e * H_ + col];
#pragma unroll
      for (int rb = 0; rb < 2; ++rb) {
#pragma unroll
        for (int r = 0; r < 8; ++r) {         // row m = rb*16 + hi*8 + r
          float vv = acc[rb][nt][r] + b1v;
          float g  = 0.5f * vv * (1.0f + erff(vv * 0.70710678118654752f)); // exact GELU
          hs[(rb * 16 + hi * 8 + r) * HS + col] = (_Float16)g;
        }
      }
    }
  }

  // ---- Phase 2: out += w[token,e] * (h @ W2 + b2)
  float wv[2][8];
#pragma unroll
  for (int rb = 0; rb < 2; ++rb)
#pragma unroll
    for (int r = 0; r < 8; ++r)
      wv[rb][r] = wts[(size_t)(tbase + rb * 16 + hi * 8 + r) * E_ + e];

  for (int oc = 0; oc < 4; ++oc) {            // 64 OUT columns per chunk
    v8f acc[2][4] = {};
    for (int kt = 0; kt < 32; ++kt) {         // K = 1024
      union { uint4 u[2]; v16h v; } A[2];
#pragma unroll
      for (int rb = 0; rb < 2; ++rb) {
        const _Float16* ap = hs + (rb * 16 + lcol) * HS + kt * 32 + hi * 16;
        A[rb].u[0] = *(const uint4*)ap;
        A[rb].u[1] = *(const uint4*)(ap + 8);
      }
#pragma unroll
      for (int nt = 0; nt < 4; ++nt) {
        union { uint4 u[2]; v16h v; } Bf;
        const _Float16* bp = w2p + ((((size_t)e * 16 + (oc * 4 + nt)) * 32 + kt) << 9) + lane * 16;
        Bf.u[0] = *(const uint4*)bp;
        Bf.u[1] = *(const uint4*)(bp + 8);
        acc[0][nt] = __builtin_amdgcn_wmma_f32_16x16x32_f16(
            false, A[0].v, false, Bf.v, (short)0, acc[0][nt], false, false);
        acc[1][nt] = __builtin_amdgcn_wmma_f32_16x16x32_f16(
            false, A[1].v, false, Bf.v, (short)0, acc[1][nt], false, false);
      }
    }
#pragma unroll
    for (int nt = 0; nt < 4; ++nt) {
      int col = oc * 64 + nt * 16 + lcol;
      float b2v = b2[e * OUT_ + col];
#pragma unroll
      for (int rb = 0; rb < 2; ++rb) {
#pragma unroll
        for (int r = 0; r < 8; ++r) {
          float vv = (acc[rb][nt][r] + b2v) * wv[rb][r];
          __hip_atomic_fetch_add(out + (size_t)(tbase + rb * 16 + hi * 8 + r) * OUT_ + col, vv,
                                 __ATOMIC_RELAXED, __HIP_MEMORY_SCOPE_AGENT);
        }
      }
    }
  }
}

// ------------------------------------------------------------------- launch
extern "C" void kernel_launch(void* const* d_in, const int* in_sizes, int n_in,
                              void* d_out, int out_size, void* d_ws, size_t ws_size,
                              hipStream_t stream) {
  const float* x      = (const float*)d_in[0];
  const float* mus    = (const float*)d_in[1];
  const float* logsig = (const float*)d_in[2];
  const float* W1     = (const float*)d_in[3];
  const float* b1     = (const float*)d_in[4];
  const float* W2     = (const float*)d_in[5];
  const float* b2     = (const float*)d_in[6];
  // d_in[7] = top_k (fixed 2 per reference)

  float* out  = (float*)d_out;                      // [8192, 256]
  float* logp = out + (size_t)NTOK * OUT_;          // [8192, 16]
  int*   topi = (int*)(logp + (size_t)NTOK * E_);   // [8192, 2]

  char*     ws  = (char*)d_ws;
  _Float16* w1p = (_Float16*)ws;                                // 8 MB packed
  _Float16* w2p = (_Float16*)(ws + (size_t)E_ * IN_ * H_ * 2);  // 8 MB packed
  float*    wts = (float*)(ws + (size_t)E_ * IN_ * H_ * 4);     // 512 KB

  zero_out_kernel<<<(NTOK * OUT_) / 256, 256, 0, stream>>>(out, NTOK * OUT_);
  pack_w_kernel<<<(E_ * (H_ / 16) * (IN_ / 32) * 32) / 256, 256, 0, stream>>>(W1, w1p, IN_, H_);
  pack_w_kernel<<<(E_ * (OUT_ / 16) * (H_ / 32) * 32) / 256, 256, 0, stream>>>(W2, w2p, H_, OUT_);
  routing_kernel<<<NTOK / 256, 256, 0, stream>>>(x, mus, logsig, wts, logp, topi);

  dim3 grid(NTOK / 64, E_);
  size_t smem = (size_t)2 * WAVE_SMEM_HALVES * sizeof(_Float16);  // 165,888 B
  moe_gemm_kernel<<<grid, 64, smem, stream>>>(x, w1p, w2p, b1, b2, wts, out);
}
